// GraphAttentionEncoder_80650895884834
// MI455X (gfx1250) — compile-verified, matrix operand order checked by reading, not run on hardware
//
#include <hip/hip_runtime.h>
#include <hip/hip_bf16.h>
#include <math.h>

typedef __attribute__((ext_vector_type(16))) __bf16          v16bf;
typedef __attribute__((ext_vector_type(8)))  float           v8f;
typedef __attribute__((ext_vector_type(4)))  float           v4f;
typedef __attribute__((ext_vector_type(8)))  unsigned short  us8;

#define NEG_SLOPE 0.2f

__device__ __forceinline__ unsigned short f2bf(float f) {
  unsigned int u = __float_as_uint(f);
  u += 0x7FFFu + ((u >> 16) & 1u);          // round-to-nearest-even
  return (unsigned short)(u >> 16);
}

union FragAB {
  v16bf v;
  unsigned short u[16];
  us8 h[2];
};

// ---------------------------------------------------------------------------
// One-shot: convert B[K,N] fp32 -> bf16, laid out in WMMA B-fragment order:
// Bsw[((kc*NT + nt)*32 + lane)*16 + j]  <-  B[kc*32 + kk(lane,j), nt*16 + (lane&15)]
// where kk = (lane>=16)*8 + (j&7) + (j>=8)*16   (ISA 7.12.2 16-bit B layout)
// ---------------------------------------------------------------------------
template <int N>
__global__ void bswizzle(const float* __restrict__ B, unsigned short* __restrict__ Bsw,
                         int K, int total) {
  const int i = blockIdx.x * blockDim.x + threadIdx.x;
  if (i >= total) return;
  const int j    = i & 15;
  const int lane = (i >> 4) & 31;
  const int nt   = (i % (N * 32)) >> 9;
  const int kc   = i / (N * 32);
  const int kk   = kc * 32 + ((lane >> 4) << 3) + (j & 7) + ((j >> 3) << 4);
  const int nn   = nt * 16 + (lane & 15);
  const float v  = (kk < K) ? B[(size_t)kk * N + nn] : 0.0f;
  Bsw[i] = f2bf(v);
}

// full A chunk, no guards: 16 fp32 per lane
// (lanes 0-15: K 0..7/16..23, lanes 16-31: +8)
__device__ __forceinline__ void load_a_fast(const float* __restrict__ arow, int k0,
                                            int half, float* __restrict__ ar) {
  const int kbase = k0 + (half << 3);
  v4f x0 = *(const v4f*)(arow + kbase);
  v4f x1 = *(const v4f*)(arow + kbase + 4);
  v4f x2 = *(const v4f*)(arow + kbase + 16);
  v4f x3 = *(const v4f*)(arow + kbase + 20);
#pragma unroll
  for (int j = 0; j < 4; ++j) {
    ar[j] = x0[j]; ar[4 + j] = x1[j]; ar[8 + j] = x2[j]; ar[12 + j] = x3[j];
  }
}

// zero-padded tail chunk (only ever the last chunk, outside the hot loop)
__device__ __forceinline__ void load_a_tail(const float* __restrict__ arow, int k0,
                                            int half, int K, float* __restrict__ ar) {
  const int kbase = k0 + (half << 3);
#pragma unroll
  for (int j = 0; j < 8; ++j) {
    int kk = kbase + j;
    ar[j] = (kk < K) ? arow[kk] : 0.0f;
    kk += 16;
    ar[8 + j] = (kk < K) ? arow[kk] : 0.0f;
  }
}

template <int NT>
__device__ __forceinline__ void load_b_frags(const us8* __restrict__ bfrag, int kc,
                                             FragAB* __restrict__ bb) {
#pragma unroll
  for (int nt = 0; nt < NT; ++nt) {
    const us8* bc = bfrag + (size_t)kc * NT * 64 + (size_t)nt * 64;
    bb[nt].h[0] = bc[0];
    bb[nt].h[1] = bc[1];
  }
}

template <int NT>
__device__ __forceinline__ void wmma_step(const float* __restrict__ ar,
                                          const FragAB* __restrict__ bb,
                                          v8f* __restrict__ acc) {
  FragAB a;
#pragma unroll
  for (int j = 0; j < 16; ++j) a.v[j] = (__bf16)ar[j];   // native cvt, RNE
#pragma unroll
  for (int nt = 0; nt < NT; ++nt)
    acc[nt] = __builtin_amdgcn_wmma_f32_16x16x32_bf16(
        false, a.v, false, bb[nt].v, (short)0, acc[nt], false, false);
}

// ---------------------------------------------------------------------------
// C[M,N] = A[M,K] (fp32 row-major) x B (pre-swizzled bf16 fragments).
// Block = 8 waves (wave32) = 256 threads -> 128 rows x N cols.
// No LDS, no barriers. NT<=4: counted, single-exit, 2x-unrolled K-loop with
// double-buffered A(raw fp32)/B(fragment) registers; guarded K-tail handled
// once after the loop so the steady state has zero branches/masks.
// ---------------------------------------------------------------------------
template <int N>
__global__ __launch_bounds__(256)
void gemm_bf16_wmma(const float* __restrict__ A, const unsigned short* __restrict__ Bsw,
                    float* __restrict__ C, int M, int K) {
  constexpr int NT = N / 16;

  const int tid  = threadIdx.x;
  const int wave = tid >> 5;
  const int lane = tid & 31;
  const int half = lane >> 4;
  const int m16  = lane & 15;

  const int waveRow0 = blockIdx.x * 128 + wave * 16;
  int arowIdx = waveRow0 + m16;
  if (arowIdx >= M) arowIdx = M - 1;        // clamp loads; stores are guarded
  const float* arow = A + (size_t)arowIdx * K;

  // this lane's B-fragment stream: 32B per tile per chunk, contiguous
  const us8* bfrag = (const us8*)Bsw + (size_t)lane * 2;

  const v8f vzero = {0.f, 0.f, 0.f, 0.f, 0.f, 0.f, 0.f, 0.f};
  v8f acc[NT];
#pragma unroll
  for (int t = 0; t < NT; ++t) acc[t] = vzero;

  const int nkf = K >> 5;                   // full 32-wide chunks
  const int rem = K & 31;

  if constexpr (NT <= 4) {
    float  ar0[16], ar1[16];
    FragAB bb0[NT], bb1[NT];
    if (nkf > 0) {
      load_a_fast(arow, 0, half, ar0);
      load_b_frags<NT>(bfrag, 0, bb0);
      int kc = 0;
      for (; kc + 2 < nkf; kc += 2) {       // steady state: no guards, no masks
        load_a_fast(arow, (kc + 1) << 5, half, ar1);
        load_b_frags<NT>(bfrag, kc + 1, bb1);
        __builtin_prefetch(arow + ((kc + 4) << 5), 0, 3);
        wmma_step<NT>(ar0, bb0, acc);
        load_a_fast(arow, (kc + 2) << 5, half, ar0);
        load_b_frags<NT>(bfrag, kc + 2, bb0);
        wmma_step<NT>(ar1, bb1, acc);
      }
      if (kc + 1 < nkf) {                   // two full chunks left
        load_a_fast(arow, (kc + 1) << 5, half, ar1);
        load_b_frags<NT>(bfrag, kc + 1, bb1);
        wmma_step<NT>(ar0, bb0, acc);
        wmma_step<NT>(ar1, bb1, acc);
      } else {                              // one full chunk left
        wmma_step<NT>(ar0, bb0, acc);
      }
    }
    if (rem) {                              // zero-padded K tail, once
      load_a_tail(arow, nkf << 5, half, K, ar0);
      load_b_frags<NT>(bfrag, nkf, bb0);
      wmma_step<NT>(ar0, bb0, acc);
    }
  } else {
    // -------- simple path (large NT: keep VGPR pressure down) --------
    for (int kc = 0; kc < nkf; ++kc) {
      float  ar[16];
      FragAB bb[NT];
      load_a_fast(arow, kc << 5, half, ar);
      load_b_frags<NT>(bfrag, kc, bb);
      wmma_step<NT>(ar, bb, acc);
    }
    if (rem) {
      float  ar[16];
      FragAB bb[NT];
      load_a_tail(arow, nkf << 5, half, K, ar);
      load_b_frags<NT>(bfrag, nkf, bb);
      wmma_step<NT>(ar, bb, acc);
    }
  }

  // C/D layout: VGPR r -> M = r + 8*(lane>=16); N = lane&15 within tile
  const int col = lane & 15;
#pragma unroll
  for (int nt = 0; nt < NT; ++nt) {
#pragma unroll
    for (int r = 0; r < 8; ++r) {
      const int row = waveRow0 + half * 8 + r;
      if (row < M) C[(size_t)row * N + nt * 16 + col] = acc[nt][r];
    }
  }
}

// ---------------------------------------------------------------------------
// elementwise / scatter kernels
// ---------------------------------------------------------------------------
__global__ void fill_f32(float* __restrict__ p, int n, float v) {
  const int i = blockIdx.x * blockDim.x + threadIdx.x;
  if (i < n) p[i] = v;
}

__global__ void deg_kernel(const int* __restrict__ dst, int E,
                           float* __restrict__ deg) {
  const int e = blockIdx.x * blockDim.x + threadIdx.x;
  if (e < E) atomicAdd(&deg[dst[e]], 1.0f);
}

__global__ void dinv_kernel(float* __restrict__ deg, int n) {
  const int i = blockIdx.x * blockDim.x + threadIdx.x;
  if (i < n) {
    const float d = deg[i];
    deg[i] = (d > 0.0f) ? rsqrtf(fmaxf(d, 1.0f)) : 0.0f;
  }
}

__global__ void gcn_agg(const int* __restrict__ src, const int* __restrict__ dst,
                        const float* __restrict__ h0, const float* __restrict__ dinv,
                        float* __restrict__ h1, int E) {
  const int t = blockIdx.x * blockDim.x + threadIdx.x;
  if (t >= E * 64) return;
  const int e = t >> 6, c = t & 63;
  const int s = src[e], d = dst[e];
  const float norm = dinv[s] * dinv[d];
  atomicAdd(&h1[(size_t)d * 64 + c], h0[(size_t)s * 64 + c] * norm);
}

__global__ void elu_kernel(float* __restrict__ h, int n) {
  const int i = blockIdx.x * blockDim.x + threadIdx.x;
  if (i < n) {
    const float v = h[i];
    h[i] = (v > 0.0f) ? v : (expf(v) - 1.0f);
  }
}

__device__ __forceinline__ void atomicMaxF(float* addr, float val) {
  const int iv = __float_as_int(val);
  if (iv >= 0) atomicMax((int*)addr, iv);
  else         atomicMin((unsigned int*)addr, (unsigned int)iv);
}

// one wave32 per edge; lane = channel within head, 4 heads of 32
__global__ __launch_bounds__(256)
void gat_logits(const int* __restrict__ src, const int* __restrict__ dst,
                const float* __restrict__ xl, const float* __restrict__ xr,
                const float* __restrict__ att, float* __restrict__ logits,
                float* __restrict__ mbuf, int E) {
  const int wave = threadIdx.x >> 5, lane = threadIdx.x & 31;
  const int e = blockIdx.x * 8 + wave;
  if (e >= E) return;
  const int s = src[e], d = dst[e];
  const float* pl = xl + (size_t)s * 128;
  const float* pr = xr + (size_t)d * 128;
#pragma unroll
  for (int h = 0; h < 4; ++h) {
    float v = pl[h * 32 + lane] + pr[h * 32 + lane];
    v = (v > 0.0f) ? v : NEG_SLOPE * v;
    float p = v * att[h * 32 + lane];
#pragma unroll
    for (int off = 16; off > 0; off >>= 1) p += __shfl_xor(p, off, 32);
    if (lane == 0) {
      logits[(size_t)e * 4 + h] = p;
      atomicMaxF(&mbuf[(size_t)d * 4 + h], p);
    }
  }
}

__global__ void gat_ex(const int* __restrict__ dst, float* __restrict__ logits,
                       const float* __restrict__ mbuf, float* __restrict__ denom,
                       int E) {
  const int t = blockIdx.x * blockDim.x + threadIdx.x;
  if (t >= E * 4) return;
  const int e = t >> 2, h = t & 3;
  const int d = dst[e];
  const float ex = __expf(logits[t] - mbuf[(size_t)d * 4 + h]);
  logits[t] = ex;                       // ex stored in place
  atomicAdd(&denom[(size_t)d * 4 + h], ex);
}

__global__ void gat_accum(const int* __restrict__ src, const int* __restrict__ dst,
                          const float* __restrict__ xl, const float* __restrict__ ex,
                          const float* __restrict__ denom, float* __restrict__ accum,
                          int E) {
  const int t = blockIdx.x * blockDim.x + threadIdx.x;
  if (t >= E * 128) return;
  const int e = t >> 7, idx = t & 127, h = idx >> 5;
  const int s = src[e], d = dst[e];
  const float alpha = ex[(size_t)e * 4 + h] / (denom[(size_t)d * 4 + h] + 1e-16f);
  atomicAdd(&accum[(size_t)d * 128 + idx], xl[(size_t)s * 128 + idx] * alpha);
}

__global__ void head_mean(const float* __restrict__ accum, float* __restrict__ out,
                          int n) {
  const int t = blockIdx.x * blockDim.x + threadIdx.x;
  if (t >= n * 32) return;
  const int node = t >> 5, c = t & 31;
  const float* p = accum + (size_t)node * 128;
  out[t] = 0.25f * (p[c] + p[32 + c] + p[64 + c] + p[96 + c]);
}

// ---------------------------------------------------------------------------
extern "C" void kernel_launch(void* const* d_in, const int* in_sizes, int n_in,
                              void* d_out, int out_size, void* d_ws, size_t ws_size,
                              hipStream_t stream) {
  const float* x   = (const float*)d_in[0];
  const int*   ei  = (const int*)d_in[1];
  const float* Wg  = (const float*)d_in[2];
  const float* Wl  = (const float*)d_in[3];
  const float* Wr  = (const float*)d_in[4];
  const float* att = (const float*)d_in[5];
  float* out = (float*)d_out;

  const int Kx = in_sizes[2] / 64;      // 2000
  const int Nn = in_sizes[0] / Kx;      // 50000
  const int E  = in_sizes[1] / 2;       // 800000
  const int* src = ei;
  const int* dst = ei + E;

  float* ws    = (float*)d_ws;
  float* h0    = ws;                            // [Nn,64]
  float* h1    = h0 + (size_t)Nn * 64;          // [Nn,64]
  float* accum = h0;                            // [Nn,128] reuses h0+h1 (dead)
  float* deg   = h1 + (size_t)Nn * 64;          // [Nn] -> becomes dinv
  float* xl    = deg + (size_t)Nn;              // [Nn,128]
  float* xr    = xl + (size_t)Nn * 128;         // [Nn,128]
  float* logit = xr + (size_t)Nn * 128;         // [E,4] -> becomes ex
  float* mbuf  = logit + (size_t)E * 4;         // [Nn,4]
  float* denom = mbuf + (size_t)Nn * 4;         // [Nn,4]

  // pre-swizzled bf16 B copies (32B-aligned), after the float workspace
  size_t foff = (size_t)(denom + (size_t)Nn * 4 - ws);
  foff = (foff + 7) & ~(size_t)7;
  const int nkg = (Kx + 31) / 32;               // 63
  const int nks = (64 + 31) / 32;               // 2
  const int totG = nkg * 64 * 32;               // swizzled W_gcn elements
  const int totS = nks * 128 * 32;              // swizzled W_l / W_r elements
  unsigned short* bswg = (unsigned short*)(ws + foff);
  unsigned short* bswl = bswg + (size_t)totG;
  unsigned short* bswr = bswl + (size_t)totS;

  const int gM = (Nn + 127) / 128;

  // 0) one-shot weight swizzles (bf16, fragment order)
  bswizzle<64><<<(totG + 255) / 256, 256, 0, stream>>>(Wg, bswg, Kx, totG);
  bswizzle<128><<<(totS + 255) / 256, 256, 0, stream>>>(Wl, bswl, 64, totS);
  bswizzle<128><<<(totS + 255) / 256, 256, 0, stream>>>(Wr, bswr, 64, totS);

  // 1) h0 = x @ W_gcn  (memory-bound on x; bf16 WMMA, fp32 accumulate)
  gemm_bf16_wmma<64><<<gM, 256, 0, stream>>>(x, bswg, h0, Nn, Kx);

  // 2) symmetric-normalized aggregation + ELU
  fill_f32<<<(Nn + 255) / 256, 256, 0, stream>>>(deg, Nn, 0.f);
  fill_f32<<<(Nn * 64 + 255) / 256, 256, 0, stream>>>(h1, Nn * 64, 0.f);
  deg_kernel<<<(E + 255) / 256, 256, 0, stream>>>(dst, E, deg);
  dinv_kernel<<<(Nn + 255) / 256, 256, 0, stream>>>(deg, Nn);
  gcn_agg<<<(E * 64 + 255) / 256, 256, 0, stream>>>(src, dst, h0, deg, h1, E);
  elu_kernel<<<(Nn * 64 + 255) / 256, 256, 0, stream>>>(h1, Nn * 64);

  // 3) xl = h @ W_l, xr = h @ W_r  (WMMA, K=64)
  gemm_bf16_wmma<128><<<gM, 256, 0, stream>>>(h1, bswl, xl, Nn, 64);
  gemm_bf16_wmma<128><<<gM, 256, 0, stream>>>(h1, bswr, xr, Nn, 64);

  // 4) GATv2 edge softmax + aggregation
  fill_f32<<<(Nn * 4 + 255) / 256, 256, 0, stream>>>(mbuf, Nn * 4, -INFINITY);
  fill_f32<<<(Nn * 4 + 255) / 256, 256, 0, stream>>>(denom, Nn * 4, 0.f);
  fill_f32<<<(Nn * 128 + 255) / 256, 256, 0, stream>>>(accum, Nn * 128, 0.f);
  gat_logits<<<(E + 7) / 8, 256, 0, stream>>>(src, dst, xl, xr, att, logit, mbuf, E);
  gat_ex<<<(E * 4 + 255) / 256, 256, 0, stream>>>(dst, logit, mbuf, denom, E);
  gat_accum<<<(E * 128 + 255) / 256, 256, 0, stream>>>(src, dst, xl, logit, denom,
                                                       accum, E);
  head_mean<<<(Nn * 32 + 255) / 256, 256, 0, stream>>>(accum, out, Nn);
}
